// AttentionReadout_83004537962836
// MI455X (gfx1250) — compile-verified, matrix operand order checked by reading, not run on hardware
//
#include <hip/hip_runtime.h>

#define HIDDEN 256
#define EPS 1e-16f

typedef _Float16 v16h __attribute__((ext_vector_type(16)));
typedef float    v8f  __attribute__((ext_vector_type(8)));

__device__ __forceinline__ unsigned enc_f32(float f) {
  unsigned u = __float_as_uint(f);
  return (u >> 31) ? ~u : (u | 0x80000000u);
}
__device__ __forceinline__ float dec_key(unsigned k) {
  unsigned u = (k >> 31) ? (k & 0x7FFFFFFFu) : ~k;
  return __uint_as_float(u);
}
__device__ __forceinline__ float silu_f(float v) {
  return v / (1.0f + __expf(-v));
}

// ---------------------------------------------------------------------------
// Pack f32 weights into f16 WMMA B-fragment order:
//   packed[((kt*NT + nt)*32 + lane)*16 + h] = w[K][n]
//   K = kt*32 + (h>>3)*16 + (lane>>4)*8 + (h&7),  n = nt*16 + (lane&15)
// ---------------------------------------------------------------------------
__global__ void k_pack(const float* __restrict__ w_v, const float* __restrict__ w_g1,
                       _Float16* __restrict__ wvP, _Float16* __restrict__ wg1P) {
  int idx = blockIdx.x * blockDim.x + threadIdx.x;
  if (idx < 256 * 256) {                       // w_v: K=256 (8 kt), N=256 (16 nt)
    int h = idx & 15, lane = (idx >> 4) & 31, t = idx >> 9;
    int nt = t & 15, kt = t >> 4;
    int K = kt * 32 + ((h >> 3) << 4) + ((lane >> 4) << 3) + (h & 7);
    int n = nt * 16 + (lane & 15);
    wvP[idx] = (_Float16)w_v[K * 256 + n];
  } else if (idx < 256 * 256 + 256 * 128) {    // w_g1: K=256 (8 kt), N=128 (8 nt)
    int i2 = idx - 256 * 256;
    int h = i2 & 15, lane = (i2 >> 4) & 31, t = i2 >> 9;
    int nt = t & 7, kt = t >> 3;
    int K = kt * 32 + ((h >> 3) << 4) + ((lane >> 4) << 3) + (h & 7);
    int n = nt * 16 + (lane & 15);
    wg1P[i2] = (_Float16)w_g1[K * 128 + n];
  }
}

__global__ void k_init(float* __restrict__ out, float* __restrict__ segsum,
                       unsigned* __restrict__ segkey, int outElems, int G) {
  int i = blockIdx.x * blockDim.x + threadIdx.x;
  if (i < outElems) out[i] = 0.0f;
  if (i < G) { segsum[i] = 0.0f; segkey[i] = 0u; }
}

// Build 16x32 f16 A-fragment from f32 row-major x (ISA 7.12.2 layout)
__device__ __forceinline__ v16h load_a_frag(const float* __restrict__ xr, int kb) {
  float4 q0 = *(const float4*)(xr + kb);
  float4 q1 = *(const float4*)(xr + kb + 4);
  float4 q2 = *(const float4*)(xr + kb + 16);
  float4 q3 = *(const float4*)(xr + kb + 20);
  v16h a;
  a[0]  = (_Float16)q0.x; a[1]  = (_Float16)q0.y; a[2]  = (_Float16)q0.z; a[3]  = (_Float16)q0.w;
  a[4]  = (_Float16)q1.x; a[5]  = (_Float16)q1.y; a[6]  = (_Float16)q1.z; a[7]  = (_Float16)q1.w;
  a[8]  = (_Float16)q2.x; a[9]  = (_Float16)q2.y; a[10] = (_Float16)q2.z; a[11] = (_Float16)q2.w;
  a[12] = (_Float16)q3.x; a[13] = (_Float16)q3.y; a[14] = (_Float16)q3.z; a[15] = (_Float16)q3.w;
  return a;
}

// ---------------------------------------------------------------------------
// Gate kernel: 64 nodes/block. hid = silu(x@w_g1+b_g1) via WMMA, then
// gate = hid@w_g2 + b_g2 fused via shuffles; atomicMax into segment key.
// ---------------------------------------------------------------------------
__global__ __launch_bounds__(256) void k_gate(
    const float* __restrict__ x, const int* __restrict__ batch,
    const _Float16* __restrict__ wg1P, const float* __restrict__ b_g1,
    const float* __restrict__ w_g2, const float* __restrict__ b_g2,
    float* __restrict__ gates, unsigned* __restrict__ segkey, int N) {
  extern __shared__ char smem[];
  _Float16* wLds   = (_Float16*)smem;            // 32768 halves = 64 KB
  float*    gateAcc = (float*)(smem + 65536);    // 64 floats

  const int tid = threadIdx.x;
  {  // stage packed w_g1 (64 KB)
    const uint4* src = (const uint4*)wg1P;
    uint4* dst = (uint4*)wLds;
#pragma unroll
    for (int i = 0; i < 16; ++i) dst[tid + i * 256] = src[tid + i * 256];
  }
  if (tid < 64) gateAcc[tid] = 0.0f;
  __syncthreads();

  const int wave = tid >> 5, lane = tid & 31;
  const int mt = wave >> 1, half = wave & 1;
  const int lanelo = lane & 15, lanehi = lane >> 4;
  const long long base = (long long)blockIdx.x * 64;
  long long row = base + mt * 16 + lanelo;
  if (row > (long long)N - 1) row = (long long)N - 1;   // clamp for loads only
  const float* xr = x + (size_t)row * HIDDEN;

  v8f acc[4];
#pragma unroll
  for (int j = 0; j < 4; ++j)
#pragma unroll
    for (int r = 0; r < 8; ++r) acc[j][r] = 0.0f;

#pragma unroll
  for (int kt = 0; kt < 8; ++kt) {
    v16h a = load_a_frag(xr, kt * 32 + lanehi * 8);
#pragma unroll
    for (int j = 0; j < 4; ++j) {
      const int nt = half * 4 + j;
      v16h b = *(const v16h*)(wLds + (((kt * 8 + nt) * 32 + lane) << 4));
      acc[j] = __builtin_amdgcn_wmma_f32_16x16x32_f16(false, a, false, b,
                                                      (short)0, acc[j], false, false);
    }
  }

  float pr[8];
#pragma unroll
  for (int r = 0; r < 8; ++r) pr[r] = 0.0f;
#pragma unroll
  for (int j = 0; j < 4; ++j) {
    const int col = (half * 4 + j) * 16 + lanelo;
    const float bg = b_g1[col];
    const float wg = w_g2[col];
#pragma unroll
    for (int r = 0; r < 8; ++r) pr[r] += silu_f(acc[j][r] + bg) * wg;
  }
#pragma unroll
  for (int r = 0; r < 8; ++r) {  // reduce across 16 lanes of each half-group
    float s = pr[r];
    s += __shfl_xor(s, 1);
    s += __shfl_xor(s, 2);
    s += __shfl_xor(s, 4);
    s += __shfl_xor(s, 8);
    if (lanelo == 0) atomicAdd(&gateAcc[mt * 16 + lanehi * 8 + r], s);
  }
  __syncthreads();
  if (tid < 64) {
    long long node = base + tid;
    if (node < N) {
      float g = gateAcc[tid] + b_g2[0];
      gates[node] = g;
      atomicMax(&segkey[batch[node]], enc_f32(g));
    }
  }
}

__global__ void k_soft(const float* __restrict__ gates, const int* __restrict__ batch,
                       const unsigned* __restrict__ segkey, float* __restrict__ evec,
                       float* __restrict__ segsum, int N) {
  int i = blockIdx.x * blockDim.x + threadIdx.x;
  if (i < N) {
    int b = batch[i];
    float ev = __expf(gates[i] - dec_key(segkey[b]));
    evec[i] = ev;
    atomicAdd(&segsum[b], ev);
  }
}

// ---------------------------------------------------------------------------
// Value kernel: 64 nodes/block, entire w_v (128 KB f16) in LDS. After the
// GEMM the weight region is reused (barrier) as a 64x256 f32 staging tile
// for a sorted-batch segmented reduction -> ~1-2 atomicAdds/col/block.
// ---------------------------------------------------------------------------
__global__ __launch_bounds__(256) void k_value(
    const float* __restrict__ x, const int* __restrict__ batch,
    const _Float16* __restrict__ wvP, const float* __restrict__ b_v,
    const float* __restrict__ evec, const float* __restrict__ segsum,
    float* __restrict__ out, int N) {
  extern __shared__ char smem[];
  _Float16* wLds    = (_Float16*)smem;                  // 65536 halves = 128 KB
  float*    redLds  = (float*)smem;                     // reuse: 64*256 f32 = 64 KB
  int*      batchLds = (int*)(smem + 131072);           // 64 ints
  float*    attnLds  = (float*)(smem + 131072 + 256);   // 64 floats

  const int tid = threadIdx.x;
  {  // stage packed w_v (128 KB)
    const uint4* src = (const uint4*)wvP;
    uint4* dst = (uint4*)wLds;
#pragma unroll
    for (int i = 0; i < 32; ++i) dst[tid + i * 256] = src[tid + i * 256];
  }
  const long long base = (long long)blockIdx.x * 64;
  if (tid < 64) {
    long long node = base + tid;
    if (node < N) {
      int b = batch[node];
      batchLds[tid] = b;
      attnLds[tid] = evec[node] / (segsum[b] + EPS);
    } else {
      batchLds[tid] = -1;
      attnLds[tid] = 0.0f;
    }
  }
  __syncthreads();

  const int wave = tid >> 5, lane = tid & 31;
  const int mt = wave >> 1, half = wave & 1;
  const int lanelo = lane & 15, lanehi = lane >> 4;
  long long row = base + mt * 16 + lanelo;
  if (row > (long long)N - 1) row = (long long)N - 1;
  const float* xr = x + (size_t)row * HIDDEN;

  v8f acc[8];
#pragma unroll
  for (int j = 0; j < 8; ++j)
#pragma unroll
    for (int r = 0; r < 8; ++r) acc[j][r] = 0.0f;

#pragma unroll
  for (int kt = 0; kt < 8; ++kt) {
    v16h a = load_a_frag(xr, kt * 32 + lanehi * 8);
#pragma unroll
    for (int j = 0; j < 8; ++j) {
      const int nt = half * 8 + j;
      v16h b = *(const v16h*)(wLds + (((kt * 16 + nt) * 32 + lane) << 4));
      acc[j] = __builtin_amdgcn_wmma_f32_16x16x32_f16(false, a, false, b,
                                                      (short)0, acc[j], false, false);
    }
  }
  __syncthreads();  // everyone done reading weights; reuse LDS as red tile

#pragma unroll
  for (int j = 0; j < 8; ++j) {
    const int col = (half * 8 + j) * 16 + lanelo;
    const float bv = b_v[col];
#pragma unroll
    for (int r = 0; r < 8; ++r) {
      const int ri = mt * 16 + lanehi * 8 + r;
      redLds[ri * 256 + col] = silu_f(acc[j][r] + bv) * attnLds[ri];
    }
  }
  __syncthreads();

  // segmented column reduction (batch is sorted)
  const int col = tid;
  float accum = 0.0f;
  int curb = -2;
#pragma unroll 4
  for (int rrow = 0; rrow < 64; ++rrow) {
    int b = batchLds[rrow];
    if (b != curb) {
      if (curb >= 0) atomicAdd(&out[(size_t)curb * 256 + col], accum);
      accum = 0.0f;
      curb = b;
    }
    if (b >= 0) accum += redLds[rrow * 256 + col];
  }
  if (curb >= 0) atomicAdd(&out[(size_t)curb * 256 + col], accum);
}

// ---------------------------------------------------------------------------
extern "C" void kernel_launch(void* const* d_in, const int* in_sizes, int n_in,
                              void* d_out, int out_size, void* d_ws, size_t ws_size,
                              hipStream_t stream) {
  const float* x    = (const float*)d_in[0];
  const int*   batch = (const int*)d_in[1];
  const float* w_g1 = (const float*)d_in[2];
  const float* b_g1 = (const float*)d_in[3];
  const float* w_g2 = (const float*)d_in[4];
  const float* b_g2 = (const float*)d_in[5];
  const float* w_v  = (const float*)d_in[6];
  const float* b_v  = (const float*)d_in[7];
  float* out = (float*)d_out;

  const int N = in_sizes[1];
  const int G = out_size / HIDDEN;

  char* ws = (char*)d_ws;
  _Float16* wvP  = (_Float16*)(ws);                              // 131072 B
  _Float16* wg1P = (_Float16*)(ws + 131072);                     // 65536 B
  float*    gates = (float*)(ws + 196608);                       // N*4
  float*    evec  = (float*)(ws + 196608 + (size_t)N * 4);       // N*4
  unsigned* segkey = (unsigned*)(ws + 196608 + (size_t)N * 8);   // G*4
  float*    segsum = (float*)(ws + 196608 + (size_t)N * 8 + (size_t)G * 4);

  (void)n_in; (void)ws_size;

  hipFuncSetAttribute((const void*)k_gate, hipFuncAttributeMaxDynamicSharedMemorySize,
                      65536 + 256);
  hipFuncSetAttribute((const void*)k_value, hipFuncAttributeMaxDynamicSharedMemorySize,
                      131072 + 512);

  k_pack<<<(256 * 256 + 256 * 128 + 255) / 256, 256, 0, stream>>>(w_v, w_g1, wvP, wg1P);
  k_init<<<(out_size + 255) / 256, 256, 0, stream>>>(out, segsum, segkey, out_size, G);

  const int nb = (N + 63) / 64;
  k_gate<<<nb, 256, 65536 + 256, stream>>>(x, batch, wg1P, b_g1, w_g2, b_g2,
                                           gates, segkey, N);
  k_soft<<<(N + 255) / 256, 256, 0, stream>>>(gates, batch, segkey, evec, segsum, N);
  k_value<<<nb, 256, 131072 + 512, stream>>>(x, batch, wvP, b_v, evec, segsum, out, N);
}
